// LSTM_85418309583143
// MI455X (gfx1250) — matrix operand order repaired and checked by reading, not compiled, not run on hardware
//
#include <hip/hip_runtime.h>
#include <math.h>

// ---------------------------------------------------------------------------
// Siamese LSTM inference for MI455X (gfx1250), bf16 WMMA path.
//
// Sizes: VOCAB=50000, EMB=256, HID=512, N=256, T=512.
// Batch rows m in [0,512): m<256 -> sequence 1, m>=256 -> sequence 2.
// Per step: gates[512,2048] = [x_t | h_prev](512x768) * Wcomb^T(2048x768) + b
// then fused LSTM cell update. 512 sequential step launches (graph-replayed).
//
// Block tile: 16 rows x 64 hidden-cols across all 4 gates. 8 waves; each wave
// owns one gate x two 16-col tiles (2 accumulators) and reuses the same LDS A
// fragment for both WMMAs per K-chunk.
// ---------------------------------------------------------------------------

typedef __attribute__((ext_vector_type(16))) __bf16 v16bf;
typedef __attribute__((ext_vector_type(8)))  __bf16 v8bf;
typedef __attribute__((ext_vector_type(8)))  float  v8f;

#define EMB   256
#define HID   512
#define NB    256           // batch per sequence
#define TSTEPS 512
#define KTOT  768           // EMB + HID
#define G4H   2048          // 4*HID
#define KCHUNKS 24          // 768 / 32

// ---- workspace layout (byte offsets) ----
#define WC_OFF   0ull                         // bf16 [2048][768]      3,145,728 B
#define BS_OFF   3145728ull                   // f32  [2048]               8,192 B
#define HB0_OFF  3153920ull                   // bf16 [512][512]         524,288 B
#define HB1_OFF  3678208ull                   // bf16 [512][512]         524,288 B
#define HF_OFF   4202496ull                   // f32  [512][512]       1,048,576 B
#define C_OFF    5251072ull                   // f32  [512][512]       1,048,576 B
// total ~6.3 MB

__device__ __forceinline__ unsigned short f2bf(float f) {
    unsigned int u = __float_as_uint(f);
    unsigned int r = u + 0x7FFFu + ((u >> 16) & 1u);   // round-to-nearest-even
    return (unsigned short)(r >> 16);
}

// branch-free sigmoid / tanh (single v_exp_f32 each, no libm range reduction)
__device__ __forceinline__ float fast_sigmoid(float x) {
    return 1.0f / (1.0f + __expf(-x));
}
__device__ __forceinline__ float fast_tanh(float x) {
    return 2.0f / (1.0f + __expf(-2.0f * x)) - 1.0f;
}

// ---------------------------------------------------------------------------
// Prep: fuse W_ih (2048x256) and W_hh (2048x512) into bf16 Wcomb[2048][768];
// bsum = b_ih + b_hh.
// ---------------------------------------------------------------------------
__global__ __launch_bounds__(256) void prep_weights(
    const float* __restrict__ Wih, const float* __restrict__ Whh,
    const float* __restrict__ bih, const float* __restrict__ bhh,
    unsigned short* __restrict__ Wc, float* __restrict__ bsum)
{
    int idx = blockIdx.x * 256 + threadIdx.x;
    if (idx < G4H * KTOT) {
        int g = idx / KTOT;
        int k = idx - g * KTOT;
        float v = (k < EMB) ? Wih[g * EMB + k] : Whh[g * HID + (k - EMB)];
        Wc[idx] = f2bf(v);
    }
    if (idx < G4H) bsum[idx] = bih[idx] + bhh[idx];
}

// zero h0 (bf16) and c0 (f32): 512*512 = 262144 elements each
__global__ __launch_bounds__(256) void init_state(
    unsigned short* __restrict__ hb0, float* __restrict__ cst)
{
    int idx = blockIdx.x * 256 + threadIdx.x;
    hb0[idx] = 0;
    cst[idx] = 0.0f;
}

// ---------------------------------------------------------------------------
// One LSTM timestep for both sequences (M = 512 batch rows).
// Grid: (8, 32) blocks of 256 threads (8 waves).
//   blockIdx.y = mt (16-row tile), blockIdx.x = jt (64 hidden-cols tile).
// Wave w: gate = w>>1, column pair = w&1 -> 16-col tiles 2*(w&1), 2*(w&1)+1.
// All 4 gates for the j-range live in this block -> cell update fuses here.
// ---------------------------------------------------------------------------
__global__ __launch_bounds__(256) void lstm_step(
    const int*  __restrict__ tok1, const int* __restrict__ tok2,
    const float* __restrict__ emb,
    const unsigned short* __restrict__ Wc,
    const float* __restrict__ bsum,
    const unsigned short* __restrict__ hin,   // h_{t-1} bf16 [512][512]
    unsigned short* __restrict__ hout,        // h_t bf16
    float* __restrict__ hf,                   // h_t f32 (for final FC)
    float* __restrict__ cst,                  // c state f32 (in/out)
    int t)
{
    __shared__ __align__(16) unsigned short As[16][KTOT + 8];  // A: [x_t|h] bf16
    __shared__ float Gs[4][16][65];                            // gate exchange

    const int tid   = threadIdx.x;
    const int jt    = blockIdx.x;          // 0..7  (64 cols each)
    const int mt    = blockIdx.y;          // 0..31
    const int mbase = mt * 16;

    // ---- Stage A into LDS: 16 rows x 768 K (embed gather + h_prev) ----
    {
        const int row = tid >> 4;          // 0..15
        const int l16 = tid & 15;          // 0..15
        const int m   = mbase + row;
        const int token = (m < NB) ? tok1[m * TSTEPS + t]
                                   : tok2[(m - NB) * TSTEPS + t];
        const float* er = emb + (size_t)token * EMB;
        #pragma unroll
        for (int k = l16 * 4; k < EMB; k += 64) {
            float4 v = *(const float4*)(er + k);
            As[row][k + 0] = f2bf(v.x);
            As[row][k + 1] = f2bf(v.y);
            As[row][k + 2] = f2bf(v.z);
            As[row][k + 3] = f2bf(v.w);
        }
        const unsigned short* hr = hin + (size_t)m * HID;
        #pragma unroll
        for (int k = l16 * 4; k < HID; k += 64) {
            *(unsigned long long*)(&As[row][EMB + k]) =
                *(const unsigned long long*)(hr + k);
        }
    }
    __syncthreads();

    // ---- WMMA: each wave two 16x16 tiles (shared A), K = 24 chunks of 32 ----
    const int wave  = tid >> 5;            // 0..7
    const int lane  = tid & 31;
    const int gidx  = wave >> 1;           // gate 0..3 (i,f,g,o)
    const int cpair = wave & 1;            // which 32-col half of the 64 j's
    const int n16   = lane & 15;           // A row (M) and B column (N)
    const int khalf = lane >> 4;           // K-half selector
    // two B columns per wave: j-cols (jt*64 + cpair*32 + {0,16}) + n16
    const int gcol0 = gidx * HID + jt * 64 + cpair * 32 + n16;
    const unsigned short* wrow0 = Wc + (size_t)gcol0 * KTOT;
    const unsigned short* wrow1 = wrow0 + (size_t)16 * KTOT;

    v8f acc0 = {};
    v8f acc1 = {};
    #pragma unroll 2
    for (int kc = 0; kc < KCHUNKS; ++kc) {
        const int K0 = kc * 32;
        // A 16x32 bf16: lane m=n16; 16 contiguous K at K0+16*khalf
        //               (same K order as B)
        const unsigned short* ap = &As[n16][K0 + 16 * khalf];
        v8bf alo = *(const v8bf*)ap;
        v8bf ahi = *(const v8bf*)(ap + 8);
        v16bf a = __builtin_shufflevector(alo, ahi,
                    0,1,2,3,4,5,6,7,8,9,10,11,12,13,14,15);
        // B 32x16 bf16: lane col=n16; 16 contiguous K at K0+16*khalf
        const unsigned short* bp0 = wrow0 + K0 + 16 * khalf;
        v8bf b0lo = *(const v8bf*)bp0;
        v8bf b0hi = *(const v8bf*)(bp0 + 8);
        v16bf b0 = __builtin_shufflevector(b0lo, b0hi,
                     0,1,2,3,4,5,6,7,8,9,10,11,12,13,14,15);
        const unsigned short* bp1 = wrow1 + K0 + 16 * khalf;
        v8bf b1lo = *(const v8bf*)bp1;
        v8bf b1hi = *(const v8bf*)(bp1 + 8);
        v16bf b1 = __builtin_shufflevector(b1lo, b1hi,
                     0,1,2,3,4,5,6,7,8,9,10,11,12,13,14,15);
        acc0 = __builtin_amdgcn_wmma_f32_16x16x32_bf16(
                   false, a, false, b0, (short)0, acc0, false, false);
        acc1 = __builtin_amdgcn_wmma_f32_16x16x32_bf16(
                   false, a, false, b1, (short)0, acc1, false, false);
    }

    // ---- exchange gate tiles through LDS ----
    // C/D layout: lane n=lane%16; VGPR r holds M = r + 8*(lane/16)
    #pragma unroll
    for (int r = 0; r < 8; ++r) {
        const int mrow = r + 8 * khalf;
        Gs[gidx][mrow][cpair * 32 + n16]      = acc0[r];
        Gs[gidx][mrow][cpair * 32 + 16 + n16] = acc1[r];
    }
    __syncthreads();

    // ---- fused LSTM cell update: 16 rows x 64 j = 1024 elements ----
    #pragma unroll
    for (int e = tid; e < 1024; e += 256) {
        const int mm = e >> 6;
        const int jj = e & 63;
        const int m  = mbase + mm;
        const int jg = jt * 64 + jj;
        float gi = Gs[0][mm][jj] + bsum[jg];
        float gf = Gs[1][mm][jj] + bsum[HID + jg];
        float gg = Gs[2][mm][jj] + bsum[2 * HID + jg];
        float go = Gs[3][mm][jj] + bsum[3 * HID + jg];
        float i_ = fast_sigmoid(gi);
        float f_ = fast_sigmoid(gf);
        float o_ = fast_sigmoid(go);
        float g_ = fast_tanh(gg);
        const size_t idx = (size_t)m * HID + jg;
        float c_ = f_ * cst[idx] + i_ * g_;
        cst[idx] = c_;
        float h_ = o_ * fast_tanh(c_);
        hf[idx]   = h_;
        hout[idx] = f2bf(h_);
    }
}

// ---------------------------------------------------------------------------
// Final: h = h1 * h2 elementwise, logits = h @ W_fc^T + b_fc, softmax.
// One block of 256 threads; thread n owns output row n. Tiny.
// ---------------------------------------------------------------------------
__global__ __launch_bounds__(256) void final_fc(
    const float* __restrict__ hf, const float* __restrict__ Wfc,
    const float* __restrict__ bfc, float* __restrict__ out)
{
    const int n = threadIdx.x;             // 0..255
    const float* h1 = hf + (size_t)n * HID;
    const float* h2 = hf + (size_t)(n + NB) * HID;
    float l0 = bfc[0], l1 = bfc[1];
    #pragma unroll 8
    for (int j = 0; j < HID; ++j) {
        float p = h1[j] * h2[j];
        l0 += p * Wfc[j];
        l1 += p * Wfc[HID + j];
    }
    float mx = fmaxf(l0, l1);
    float e0 = __expf(l0 - mx);
    float e1 = __expf(l1 - mx);
    float s  = e0 + e1;
    out[n * 2 + 0] = e0 / s;
    out[n * 2 + 1] = e1 / s;
}

// ---------------------------------------------------------------------------
extern "C" void kernel_launch(void* const* d_in, const int* in_sizes, int n_in,
                              void* d_out, int out_size, void* d_ws, size_t ws_size,
                              hipStream_t stream)
{
    const int*   tok1 = (const int*)  d_in[0];   // [256,512]
    const int*   tok2 = (const int*)  d_in[1];   // [256,512]
    const float* emb  = (const float*)d_in[2];   // [50000,256]
    const float* Wih  = (const float*)d_in[3];   // [2048,256]
    const float* Whh  = (const float*)d_in[4];   // [2048,512]
    const float* bih  = (const float*)d_in[5];   // [2048]
    const float* bhh  = (const float*)d_in[6];   // [2048]
    const float* Wfc  = (const float*)d_in[7];   // [2,512]
    const float* bfc  = (const float*)d_in[8];   // [2]
    float* out = (float*)d_out;                  // [256,2]

    char* ws = (char*)d_ws;
    unsigned short* Wc   = (unsigned short*)(ws + WC_OFF);
    float*          bsum = (float*)         (ws + BS_OFF);
    unsigned short* hb0  = (unsigned short*)(ws + HB0_OFF);
    unsigned short* hb1  = (unsigned short*)(ws + HB1_OFF);
    float*          hf   = (float*)         (ws + HF_OFF);
    float*          cst  = (float*)         (ws + C_OFF);

    // weights -> bf16 fused matrix; bias sum
    prep_weights<<<(G4H * KTOT + 255) / 256, 256, 0, stream>>>(
        Wih, Whh, bih, bhh, Wc, bsum);
    // h0 = 0 (bf16), c0 = 0
    init_state<<<(2 * NB * HID) / 256, 256, 0, stream>>>(hb0, cst);

    // 512 sequential timesteps; h ping-pongs between hb0/hb1
    dim3 grid(8, 32);
    for (int t = 0; t < TSTEPS; ++t) {
        unsigned short* hin  = (t & 1) ? hb1 : hb0;
        unsigned short* hout = (t & 1) ? hb0 : hb1;
        lstm_step<<<grid, 256, 0, stream>>>(
            tok1, tok2, emb, Wc, bsum, hin, hout, hf, cst, t);
    }

    // final FC + softmax from f32 h of the last step
    final_fc<<<1, 256, 0, stream>>>(hf, Wfc, bfc, out);
}
